// SRNN_67138928771245
// MI455X (gfx1250) — matrix operand order repaired and checked by reading, not compiled, not run on hardware
//
#include <hip/hip_runtime.h>
#include <hip/hip_bf16.h>
#include <math.h>

#define BB 64
#define TT 512
#define XX 64
#define HH 512
#define ZZ 128
#define RR 1024
#define GG 3072   // 3R
#define KA 1088   // X+R
#define KQ 1152   // R+Z

#define GRU_BLOCKS 96
#define SCAN_BLOCKS 32

typedef __bf16 bf16;
typedef __attribute__((ext_vector_type(16))) __bf16 v16bf;
typedef __attribute__((ext_vector_type(8)))  __bf16 v8bf;
typedef __attribute__((ext_vector_type(8)))  float  v8f;

static __device__ __forceinline__ v8f wmma_bf16(v16bf a, v16bf b, v8f c) {
    // emits v_wmma_f32_16x16x32_bf16
    return __builtin_amdgcn_wmma_f32_16x16x32_bf16(false, a, false, b, (short)0, c,
                                                   false, false);
}

// A fragment (16x32 bf16, row-major, leading dim lda).
// Layout: lanes 0-15 hold row M=lane, K 0-15;
// lanes 16-31 hold row M=lane-16, K 16-31 (matches load_b_frag).
static __device__ __forceinline__ v16bf load_a_frag(const bf16* __restrict__ base,
                                                    int lda, int lane) {
    const int r  = lane & 15;
    const int kb = (lane >> 4) << 4;            // 0 or 16
    const bf16* p = base + (size_t)r * lda + kb;
    v8bf lo = *(const v8bf*)(p);
    v8bf hi = *(const v8bf*)(p + 8);
    v16bf a;
#pragma unroll
    for (int i = 0; i < 8; ++i) { a[i] = lo[i]; a[i + 8] = hi[i]; }
    return a;
}

// B fragment (32x16 bf16) built from weight W[N,K] row-major (so B[k][n]=W[n][k]).
// Lane holds col n=lane&15 for K = (lane>>4)*16 .. +15 -> 16 contiguous bf16.
static __device__ __forceinline__ v16bf load_b_frag(const bf16* __restrict__ Wbase,
                                                    int ldw, int lane) {
    const int n  = lane & 15;
    const int kg = (lane >> 4) << 4;            // 0 or 16
    return *(const v16bf*)(Wbase + (size_t)n * ldw + kg);
}

// Block-cooperative copy of a 16-row A panel (row length Kdim, contiguous rows)
// into LDS with padded row stride srow (bank-conflict-free ds_load_b128 later).
static __device__ __forceinline__ void stage_rows(bf16* __restrict__ sA,
                                                  const bf16* __restrict__ src,
                                                  int Kdim, int srow) {
    const int chunks = Kdim >> 3;               // v8bf chunks per row
    for (int i = threadIdx.x; i < 16 * chunks; i += blockDim.x) {
        const int r = i / chunks, cidx = i - r * chunks;
        ((v8bf*)(sA + (size_t)r * srow))[cidx] =
            ((const v8bf*)(src + (size_t)r * Kdim))[cidx];
    }
}

// Agent-scope generation barrier for persistent cooperative kernels.
static __device__ __forceinline__ void grid_sync(unsigned* bar, unsigned nblocks,
                                                 unsigned gen) {
    __syncthreads();
    if (threadIdx.x == 0) {
        __threadfence();
        __hip_atomic_fetch_add(bar, 1u, __ATOMIC_RELEASE, __HIP_MEMORY_SCOPE_AGENT);
        while (__hip_atomic_load(bar, __ATOMIC_ACQUIRE, __HIP_MEMORY_SCOPE_AGENT)
               < gen * nblocks) {
            __builtin_amdgcn_s_sleep(2);
        }
    }
    __syncthreads();
}

static __device__ __forceinline__ float sigmoidf_(float x) {
    return 1.0f / (1.0f + __expf(-x));
}
static __device__ __forceinline__ float softplusf_(float x) {
    return (x > 20.0f) ? x : log1pf(__expf(x));
}
static __device__ __forceinline__ float lrelu_(float x) {
    return x > 0.0f ? x : 0.01f * x;
}

// ---------------------------------------------------------------------------
// Elementwise prep kernels
// ---------------------------------------------------------------------------
__global__ void f32_to_bf16_kernel(const float* __restrict__ src,
                                   bf16* __restrict__ dst, int n) {
    int i = blockIdx.x * blockDim.x + threadIdx.x;
    if (i < n) dst[i] = (bf16)src[i];
}

// builds u[t,b,x] = (t==0)?0:x[b,t-1,x] and x_enc[t,b,x] = x[b,t,x], both bf16
__global__ void build_inputs_kernel(const float* __restrict__ x,
                                    bf16* __restrict__ u, bf16* __restrict__ xe) {
    int i = blockIdx.x * blockDim.x + threadIdx.x;
    if (i >= TT * BB * XX) return;
    int t = i / (BB * XX);
    int r = i - t * BB * XX;
    int b = r / XX, c = r - b * XX;
    xe[i] = (bf16)x[((size_t)b * TT + t) * XX + c];
    u[i]  = (t == 0) ? (bf16)0.0f : (bf16)x[((size_t)b * TT + (t - 1)) * XX + c];
}

__global__ void init_kernel(unsigned* __restrict__ bars, float* __restrict__ lp,
                            float* __restrict__ kl) {
    int i = threadIdx.x;
    if (i < 8) bars[i] = 0u;
    if (i < BB) { lp[i] = 0.0f; kl[i] = 0.0f; }
}

__global__ void finalize_kernel(const float* __restrict__ lp,
                                const float* __restrict__ kl,
                                float* __restrict__ out) {
    int i = threadIdx.x;
    if (i < BB) out[i] = lp[i] - kl[i];
}

// ---------------------------------------------------------------------------
// Generic WMMA GEMM: C[M,N] = A[M,K]_bf16 @ W[N,K]_bf16^T + bias, one 16x16
// output tile per wave. Output bf16. (Fully parallel one-pass GEMMs keep the
// global path: no reuse window worth staging.)
// ---------------------------------------------------------------------------
__global__ void __launch_bounds__(256)
gemm_bias_kernel(const bf16* __restrict__ A, int lda,
                 const bf16* __restrict__ W, int ldw,
                 const float* __restrict__ bias,
                 bf16* __restrict__ Cb, int ldc, int M, int N, int K) {
    const int lane = threadIdx.x & 31;
    const int wave = threadIdx.x >> 5;
    const int tile = blockIdx.x * 8 + wave;
    const int ntn  = N >> 4;
    const int mt   = tile / ntn;
    const int nt   = tile - mt * ntn;
    if (mt >= (M >> 4)) return;  // wave-uniform: EXEC stays all-ones for WMMA
    const bf16* Ab = A + (size_t)(mt << 4) * lda;
    const bf16* Wb = W + (size_t)(nt << 4) * ldw;
    v8f c = {};
    for (int k = 0; k < K; k += 32) {
        __builtin_prefetch(Wb + (size_t)k + 128, 0, 1);
        v16bf a = load_a_frag(Ab + k, lda, lane);
        v16bf b = load_b_frag(Wb + k, ldw, lane);
        c = wmma_bf16(a, b, c);
    }
    const int col = lane & 15;
    const int rb  = (lane >> 4) << 3;
    const float bv = bias[(nt << 4) + col];
#pragma unroll
    for (int i = 0; i < 8; ++i)
        Cb[(size_t)((mt << 4) + rb + i) * ldc + (nt << 4) + col] = (bf16)(c[i] + bv);
}

// ---------------------------------------------------------------------------
// Concat + (optional time-reversed) GEMM over M = T*B rows (m = t*B + b).
// A[m, k] = k < K1 ? S1[tsel, b, k] : S2[tsel, b, k-K1], tsel = rev? T-1-t : t.
// K1 % 32 == 0 so a 32-wide K-tile never straddles the boundary.
// ---------------------------------------------------------------------------
__global__ void __launch_bounds__(256)
gemm_concat_kernel(const bf16* __restrict__ S1, int K1, int ld1,
                   const bf16* __restrict__ S2, int ld2,
                   const bf16* __restrict__ W, int ldw,
                   const float* __restrict__ bias,
                   bf16* __restrict__ Cb, int ldc, int N, int K, int reverse) {
    const int lane = threadIdx.x & 31;
    const int wave = threadIdx.x >> 5;
    const int tile = blockIdx.x * 8 + wave;
    const int ntn  = N >> 4;
    const int mt   = tile / ntn;
    const int nt   = tile - mt * ntn;
    if (mt >= ((TT * BB) >> 4)) return;
    const int m0 = mt << 4;
    const int t  = m0 / BB;
    const int b0 = m0 - t * BB;          // 16 rows share one t (BB % 16 == 0)
    const int ts = reverse ? (TT - 1 - t) : t;
    const bf16* A1 = S1 + ((size_t)ts * BB + b0) * ld1;
    const bf16* A2 = S2 + ((size_t)ts * BB + b0) * ld2;
    const bf16* Wb = W + (size_t)(nt << 4) * ldw;
    v8f c = {};
    for (int k = 0; k < K; k += 32) {
        __builtin_prefetch(Wb + (size_t)k + 128, 0, 1);
        v16bf a = (k < K1) ? load_a_frag(A1 + k, ld1, lane)
                           : load_a_frag(A2 + (k - K1), ld2, lane);
        v16bf b = load_b_frag(Wb + k, ldw, lane);
        c = wmma_bf16(a, b, c);
    }
    const int col = lane & 15;
    const int rb  = (lane >> 4) << 3;
    const float bv = bias[(nt << 4) + col];
#pragma unroll
    for (int i = 0; i < 8; ++i)
        Cb[(size_t)(m0 + rb + i) * ldc + (nt << 4) + col] = (bf16)(c[i] + bv);
}

// ---------------------------------------------------------------------------
// Persistent GRU recurrence. Per step t:
//   stage:  block's shared 16-row h panel -> LDS (ds_store_b128, padded rows)
//   phase1: hg[64,3072] = h @ Wh^T + bh (768 WMMA tiles == 768 waves; A from LDS)
//   phase2: fused PyTorch-GRU gate math, fp32 state + bf16 mirror,
//           store state into out_bf16 (shifted for 'd', reversed for 'a').
// All 8 waves of a block share one M-tile (192 % 8 == 0), so LDS staging cuts
// redundant L2 A-traffic ~8x on the sequential critical path.
// ---------------------------------------------------------------------------
__global__ void __launch_bounds__(256)
gru_kernel(const bf16* __restrict__ xg /*[T,B,3R] bf16, bias included*/,
           const bf16* __restrict__ Wh /*[3R,R]*/, const float* __restrict__ bh,
           bf16* __restrict__ out_bf16 /*[T,B,R]*/,
           float* __restrict__ h_f32, bf16* __restrict__ h_bf16,
           float* __restrict__ hg /*[B,3R]*/, unsigned* __restrict__ bar,
           int reverse_mode) {
    __shared__ bf16 sH[16 * (RR + 8)];     // padded: conflict-free ds_load_b128
    const int tid      = blockIdx.x * blockDim.x + threadIdx.x;
    const int nthreads = GRU_BLOCKS * 256;
    unsigned gen = 0;

    for (int i = tid; i < BB * RR; i += nthreads) {
        h_f32[i]  = 0.0f;
        h_bf16[i] = (bf16)0.0f;
        if (!reverse_mode) out_bf16[i] = (bf16)0.0f;  // d[0] = h0 = 0
    }
    grid_sync(bar, GRU_BLOCKS, ++gen);

    const int lane = threadIdx.x & 31;
    const int wave = threadIdx.x >> 5;
    const int tile = blockIdx.x * 8 + wave;        // 0..767
    const int mtb  = (blockIdx.x * 8) / (GG >> 4); // block-uniform M-tile
    const int nt   = tile - mtb * (GG >> 4);
    const bf16* Wb = Wh + (size_t)(nt << 4) * RR;
    const int col = lane & 15, rb = (lane >> 4) << 3, n = (nt << 4) + col;
    const float bv = bh[n];

    for (int t = 0; t < TT; ++t) {
        // stage shared h panel, then recurrent GEMM
        stage_rows(sH, h_bf16 + (size_t)(mtb << 4) * RR, RR, RR + 8);
        __syncthreads();
        {
            v8f c = {};
            for (int k = 0; k < RR; k += 32) {
                v16bf a = load_a_frag(sH + k, RR + 8, lane);   // ds_load path
                v16bf b = load_b_frag(Wb + k, RR, lane);
                c = wmma_bf16(a, b, c);
            }
#pragma unroll
            for (int i = 0; i < 8; ++i)
                hg[(size_t)((mtb << 4) + rb + i) * GG + n] = c[i] + bv;
        }
        grid_sync(bar, GRU_BLOCKS, ++gen);

        // phase 2: gate math
        const bf16* xgt = xg + (size_t)t * BB * GG;
        for (int i = tid; i < BB * RR; i += nthreads) {
            const int b = i >> 10, j = i & (RR - 1);
            const size_t o = (size_t)b * GG + j;
            const float xr = (float)xgt[o];
            const float xz = (float)xgt[o + RR];
            const float xn = (float)xgt[o + 2 * RR];
            const float hr = hg[o];
            const float hz = hg[o + RR];
            const float hn = hg[o + 2 * RR];
            const float hold = h_f32[i];
            const float r  = sigmoidf_(xr + hr);
            const float u  = sigmoidf_(xz + hz);
            const float nn = tanhf(xn + r * hn);
            const float hnew = (1.0f - u) * nn + u * hold;
            h_f32[i]  = hnew;
            h_bf16[i] = (bf16)hnew;
            if (reverse_mode) {
                out_bf16[(size_t)(TT - 1 - t) * BB * RR + i] = (bf16)hnew;  // a[t]
            } else if (t < TT - 1) {
                out_bf16[(size_t)(t + 1) * BB * RR + i] = (bf16)hnew;       // d shifted
            }
        }
        grid_sync(bar, GRU_BLOCKS, ++gen);
    }
}

// ---------------------------------------------------------------------------
// Latent-scan MLP layers with LDS-staged A panels.
// Block<->tile alignment: all 8 waves of a block share (net, mt), so the A
// panel is staged once per block. net 0 = p (prior), net 1 = q (posterior).
// ---------------------------------------------------------------------------
static __device__ __forceinline__ void mlp_layer_lds(int gw, int lane,
        const bf16* __restrict__ sA, int Kdim, int srow,
        const bf16* __restrict__ W0, const float* __restrict__ b0,
        const bf16* __restrict__ W1, const float* __restrict__ b1,
        bf16* __restrict__ out /* [2][64][HH] */) {
    const int ntn = HH >> 4;                 // 32
    const int tpn = (BB >> 4) * ntn;         // 128
    if (gw >= 2 * tpn) return;
    const int net = gw / tpn;
    const int sub = gw - net * tpn;
    const int mt  = sub / ntn, nt = sub - mt * ntn;
    const bf16* W = (net ? W1 : W0) + (size_t)(nt << 4) * Kdim;
    const float* bias = net ? b1 : b0;
    v8f c = {};
    for (int k = 0; k < Kdim; k += 32) {
        v16bf a = load_a_frag(sA + k, srow, lane);
        v16bf b = load_b_frag(W + k, Kdim, lane);
        c = wmma_bf16(a, b, c);
    }
    const int col = lane & 15, rb = (lane >> 4) << 3, n = (nt << 4) + col;
    const float bv = bias[n];
#pragma unroll
    for (int i = 0; i < 8; ++i)
        out[((size_t)net * BB + (mt << 4) + rb + i) * HH + n] =
            (bf16)lrelu_(c[i] + bv);
}

static __device__ __forceinline__ void gate_layer_lds(int gw, int lane,
        const bf16* __restrict__ sA, int srow,
        const bf16* __restrict__ W0, const float* __restrict__ b0,
        const bf16* __restrict__ W1, const float* __restrict__ b1,
        float* __restrict__ out /* [2][64][2Z] */) {
    const int Nout = 2 * ZZ;                 // 256
    const int ntn  = Nout >> 4;              // 16
    const int tpn  = (BB >> 4) * ntn;        // 64
    if (gw >= 2 * tpn) return;
    const int net = gw / tpn;
    const int sub = gw - net * tpn;
    const int mt  = sub / ntn, nt = sub - mt * ntn;
    const bf16* W = (net ? W1 : W0) + (size_t)(nt << 4) * HH;
    const float* bias = net ? b1 : b0;
    v8f c = {};
    for (int k = 0; k < HH; k += 32) {
        v16bf a = load_a_frag(sA + k, srow, lane);
        v16bf b = load_b_frag(W + k, HH, lane);
        c = wmma_bf16(a, b, c);
    }
    const int col = lane & 15, rb = (lane >> 4) << 3, n = (nt << 4) + col;
    const float bv = bias[n];
#pragma unroll
    for (int i = 0; i < 8; ++i)
        out[((size_t)net * BB + (mt << 4) + rb + i) * Nout + n] = c[i] + bv;
}

// ---------------------------------------------------------------------------
// Persistent latent scan: prior/posterior MLPs, rsample, KL accumulation.
// ---------------------------------------------------------------------------
__global__ void __launch_bounds__(256)
scan_kernel(const bf16* __restrict__ d_bf, const bf16* __restrict__ a_bf,
            const float* __restrict__ eps /*[T,B,Z]*/,
            const bf16* Wp1, const float* bp1, const bf16* Wp2, const float* bp2,
            const bf16* Wp3, const float* bp3, const bf16* Wpg, const float* bpg,
            const bf16* Wq1, const float* bq1, const bf16* Wq2, const float* bq2,
            const bf16* Wq3, const float* bq3, const bf16* Wqg, const float* bqg,
            bf16* __restrict__ inbuf /*[2][64][KQ]*/,
            bf16* __restrict__ hA /*[2][64][HH]*/, bf16* __restrict__ hB,
            float* __restrict__ obuf /*[2][64][2Z]*/,
            bf16* __restrict__ zprev /*[64][Z]*/,
            bf16* __restrict__ z_out /*[T,B,Z]*/, float* __restrict__ kl /*[64]*/,
            unsigned* __restrict__ bar) {
    __shared__ bf16 sA[16 * (KQ + 8)];      // max panel: layer-1 K = 1152
    const int tid      = blockIdx.x * blockDim.x + threadIdx.x;
    const int nthreads = SCAN_BLOCKS * 256;
    unsigned gen = 0;
    for (int i = tid; i < BB * ZZ; i += nthreads) zprev[i] = (bf16)0.0f;  // z0 = 0
    grid_sync(bar, SCAN_BLOCKS, ++gen);

    const int lane = threadIdx.x & 31;
    const int wave = threadIdx.x >> 5;
    const int gw   = blockIdx.x * 8 + wave;   // 0..255
    const int gw0  = blockIdx.x * 8;          // block-uniform tile base

    for (int t = 0; t < TT; ++t) {
        // phase A: inbuf[net][b] = concat(net? a_t : d_t, z_prev)
        for (int i = tid; i < 2 * BB * KQ; i += nthreads) {
            const int net = i / (BB * KQ);
            const int rem = i - net * BB * KQ;
            const int b = rem / KQ, k = rem - b * KQ;
            bf16 v;
            if (k < RR)
                v = (net ? a_bf : d_bf)[(size_t)t * BB * RR + (size_t)b * RR + k];
            else
                v = zprev[b * ZZ + (k - RR)];
            inbuf[i] = v;
        }
        grid_sync(bar, SCAN_BLOCKS, ++gen);
        // layer 1 (K = KQ)
        {
            const int net0 = gw0 >> 7, mt0 = (gw0 & 127) >> 5;
            stage_rows(sA, inbuf + ((size_t)net0 * BB + (mt0 << 4)) * KQ, KQ, KQ + 8);
            __syncthreads();
            mlp_layer_lds(gw, lane, sA, KQ, KQ + 8, Wp1, bp1, Wq1, bq1, hA);
        }
        grid_sync(bar, SCAN_BLOCKS, ++gen);
        // layer 2 (K = HH)
        {
            const int net0 = gw0 >> 7, mt0 = (gw0 & 127) >> 5;
            stage_rows(sA, hA + ((size_t)net0 * BB + (mt0 << 4)) * HH, HH, HH + 8);
            __syncthreads();
            mlp_layer_lds(gw, lane, sA, HH, HH + 8, Wp2, bp2, Wq2, bq2, hB);
        }
        grid_sync(bar, SCAN_BLOCKS, ++gen);
        // layer 3 (K = HH)
        {
            const int net0 = gw0 >> 7, mt0 = (gw0 & 127) >> 5;
            stage_rows(sA, hB + ((size_t)net0 * BB + (mt0 << 4)) * HH, HH, HH + 8);
            __syncthreads();
            mlp_layer_lds(gw, lane, sA, HH, HH + 8, Wp3, bp3, Wq3, bq3, hA);
        }
        grid_sync(bar, SCAN_BLOCKS, ++gen);
        // gauss gate (Nout = 2Z; only gw < 128 active, block-uniform)
        {
            if (gw0 < 128) {
                const int net0 = gw0 >> 6, mt0 = (gw0 & 63) >> 4;
                stage_rows(sA, hA + ((size_t)net0 * BB + (mt0 << 4)) * HH, HH, HH + 8);
            }
            __syncthreads();
            gate_layer_lds(gw, lane, sA, HH + 8, Wpg, bpg, Wqg, bqg, obuf);
        }
        grid_sync(bar, SCAN_BLOCKS, ++gen);
        // phase F: rsample + KL
        for (int i = tid; i < BB * ZZ; i += nthreads) {
            const int b = i >> 7, zj = i & (ZZ - 1);
            const float pmu  = obuf[(size_t)b * 2 * ZZ + zj];
            const float ppre = obuf[(size_t)b * 2 * ZZ + ZZ + zj];
            const float qmu  = obuf[((size_t)BB + b) * 2 * ZZ + zj];
            const float qpre = obuf[((size_t)BB + b) * 2 * ZZ + ZZ + zj];
            const float psd = softplusf_(ppre);
            const float qsd = softplusf_(qpre);
            const float e   = eps[((size_t)t * BB + b) * ZZ + zj];
            const float zv  = qmu + qsd * e;
            const bf16 zb   = (bf16)zv;
            zprev[i] = zb;
            z_out[((size_t)t * BB + b) * ZZ + zj] = zb;
            const float dm  = qmu - pmu;
            const float klc = logf(psd) - logf(qsd)
                            + (qsd * qsd + dm * dm) / (2.0f * psd * psd) - 0.5f;
            atomicAdd(&kl[b], klc);
        }
        grid_sync(bar, SCAN_BLOCKS, ++gen);
    }
}

// ---------------------------------------------------------------------------
// Decoder GEMM fused with Gaussian log-prob reduction (mu_x never stored).
// ---------------------------------------------------------------------------
__global__ void __launch_bounds__(256)
decoder_logprob_kernel(const bf16* __restrict__ z_bf, const bf16* __restrict__ d_bf,
                       const bf16* __restrict__ Wd /*[X,KQ]*/,
                       const float* __restrict__ bd,
                       const float* __restrict__ x /*[B,T,X]*/,
                       float* __restrict__ lp /*[B] accum*/) {
    const float LOG2PI = 1.8378770664093453f;
    const int lane = threadIdx.x & 31;
    const int wave = threadIdx.x >> 5;
    const int tile = blockIdx.x * 8 + wave;
    const int ntn  = XX >> 4;  // 4
    const int mt   = tile / ntn;
    const int nt   = tile - mt * ntn;
    if (mt >= ((TT * BB) >> 4)) return;
    const int m0 = mt << 4;
    const int t  = m0 / BB, b0 = m0 - t * BB;
    const bf16* A1 = z_bf + ((size_t)t * BB + b0) * ZZ;
    const bf16* A2 = d_bf + ((size_t)t * BB + b0) * RR;
    const bf16* Wb = Wd + (size_t)(nt << 4) * KQ;
    v8f c = {};
    for (int k = 0; k < KQ; k += 32) {
        v16bf a = (k < ZZ) ? load_a_frag(A1 + k, ZZ, lane)
                           : load_a_frag(A2 + (k - ZZ), RR, lane);
        v16bf b = load_b_frag(Wb + k, KQ, lane);
        c = wmma_bf16(a, b, c);
    }
    const int col = lane & 15, rb = (lane >> 4) << 3;
    const int n = (nt << 4) + col;
    const float bv = bd[n];
#pragma unroll
    for (int i = 0; i < 8; ++i) {
        const int b  = b0 + rb + i;
        const float mu = c[i] + bv;
        const float xv = x[((size_t)b * TT + t) * XX + n];
        const float e  = xv - mu;
        atomicAdd(&lp[b], -0.5f * (e * e + LOG2PI));
    }
}

// ---------------------------------------------------------------------------
// Host launcher
// ---------------------------------------------------------------------------
extern "C" void kernel_launch(void* const* d_in, const int* in_sizes, int n_in,
                              void* d_out, int out_size, void* d_ws, size_t ws_size,
                              hipStream_t stream) {
    const float* x    = (const float*)d_in[0];
    const float* eps  = (const float*)d_in[1];
    const float* Wi_d = (const float*)d_in[2];
    const float* Wh_d = (const float*)d_in[3];
    const float* bi_d = (const float*)d_in[4];
    const float* bh_d = (const float*)d_in[5];
    const float* Wi_a = (const float*)d_in[6];
    const float* Wh_a = (const float*)d_in[7];
    const float* bi_a = (const float*)d_in[8];
    const float* bh_a = (const float*)d_in[9];
    const float* Wq1 = (const float*)d_in[10]; const float* bq1 = (const float*)d_in[11];
    const float* Wq2 = (const float*)d_in[12]; const float* bq2 = (const float*)d_in[13];
    const float* Wq3 = (const float*)d_in[14]; const float* bq3 = (const float*)d_in[15];
    const float* Wqg = (const float*)d_in[16]; const float* bqg = (const float*)d_in[17];
    const float* Wp1 = (const float*)d_in[18]; const float* bp1 = (const float*)d_in[19];
    const float* Wp2 = (const float*)d_in[20]; const float* bp2 = (const float*)d_in[21];
    const float* Wp3 = (const float*)d_in[22]; const float* bp3 = (const float*)d_in[23];
    const float* Wpg = (const float*)d_in[24]; const float* bpg = (const float*)d_in[25];
    const float* Wd  = (const float*)d_in[26]; const float* bd  = (const float*)d_in[27];
    // d_in[28] = x_sl (all == T; mask is all-ones)
    float* out = (float*)d_out;

    char* ws = (char*)d_ws;
    size_t off = 0;
    auto alloc = [&](size_t bytes) -> char* {
        char* p = ws + off;
        off += (bytes + 255) & ~(size_t)255;
        return p;
    };

    // bf16 weight mirrors
    bf16* bWi_d = (bf16*)alloc((size_t)GG * XX * 2);
    bf16* bWh_d = (bf16*)alloc((size_t)GG * RR * 2);
    bf16* bWi_a = (bf16*)alloc((size_t)GG * KA * 2);
    bf16* bWh_a = (bf16*)alloc((size_t)GG * RR * 2);
    bf16* bWq1  = (bf16*)alloc((size_t)HH * KQ * 2);
    bf16* bWq2  = (bf16*)alloc((size_t)HH * HH * 2);
    bf16* bWq3  = (bf16*)alloc((size_t)HH * HH * 2);
    bf16* bWqg  = (bf16*)alloc((size_t)2 * ZZ * HH * 2);
    bf16* bWp1  = (bf16*)alloc((size_t)HH * KQ * 2);
    bf16* bWp2  = (bf16*)alloc((size_t)HH * HH * 2);
    bf16* bWp3  = (bf16*)alloc((size_t)HH * HH * 2);
    bf16* bWpg  = (bf16*)alloc((size_t)2 * ZZ * HH * 2);
    bf16* bWd   = (bf16*)alloc((size_t)XX * KQ * 2);
    // activations
    bf16* u_bf  = (bf16*)alloc((size_t)TT * BB * XX * 2);
    bf16* xe_bf = (bf16*)alloc((size_t)TT * BB * XX * 2);
    bf16* xg_bf = (bf16*)alloc((size_t)TT * BB * GG * 2);  // reused by both GRUs
    bf16* d_bf  = (bf16*)alloc((size_t)TT * BB * RR * 2);
    bf16* a_bf  = (bf16*)alloc((size_t)TT * BB * RR * 2);
    bf16* z_bf  = (bf16*)alloc((size_t)TT * BB * ZZ * 2);
    // GRU scratch
    float* hg_f = (float*)alloc((size_t)BB * GG * 4);
    float* h_f  = (float*)alloc((size_t)BB * RR * 4);
    bf16*  h_b  = (bf16*)alloc((size_t)BB * RR * 2);
    // scan scratch
    bf16*  inbuf = (bf16*)alloc((size_t)2 * BB * KQ * 2);
    bf16*  hA    = (bf16*)alloc((size_t)2 * BB * HH * 2);
    bf16*  hB    = (bf16*)alloc((size_t)2 * BB * HH * 2);
    float* obuf  = (float*)alloc((size_t)2 * BB * 2 * ZZ * 4);
    bf16*  zprev = (bf16*)alloc((size_t)BB * ZZ * 2);
    float* lp    = (float*)alloc((size_t)BB * 4);
    float* kl    = (float*)alloc((size_t)BB * 4);
    unsigned* bars = (unsigned*)alloc(8 * 4);

    init_kernel<<<1, 256, 0, stream>>>(bars, lp, kl);

    auto cvt = [&](const float* src, bf16* dst, int n) {
        f32_to_bf16_kernel<<<(n + 255) / 256, 256, 0, stream>>>(src, dst, n);
    };
    cvt(Wi_d, bWi_d, GG * XX);
    cvt(Wh_d, bWh_d, GG * RR);
    cvt(Wi_a, bWi_a, GG * KA);
    cvt(Wh_a, bWh_a, GG * RR);
    cvt(Wq1, bWq1, HH * KQ); cvt(Wq2, bWq2, HH * HH);
    cvt(Wq3, bWq3, HH * HH); cvt(Wqg, bWqg, 2 * ZZ * HH);
    cvt(Wp1, bWp1, HH * KQ); cvt(Wp2, bWp2, HH * HH);
    cvt(Wp3, bWp3, HH * HH); cvt(Wpg, bWpg, 2 * ZZ * HH);
    cvt(Wd, bWd, XX * KQ);

    build_inputs_kernel<<<(TT * BB * XX + 255) / 256, 256, 0, stream>>>(x, u_bf, xe_bf);

    // xg_d = u @ Wi_d^T + bi_d  (M=T*B, K=X, N=3R)
    {
        const int M = TT * BB, N = GG;
        const int blocks = ((M >> 4) * (N >> 4) + 7) / 8;
        gemm_bias_kernel<<<blocks, 256, 0, stream>>>(u_bf, XX, bWi_d, XX, bi_d,
                                                     xg_bf, GG, M, N, XX);
    }
    // forward deterministic GRU -> d (shifted so d_t precedes x_t)
    gru_kernel<<<GRU_BLOCKS, 256, 0, stream>>>(xg_bf, bWh_d, bh_d, d_bf,
                                               h_f, h_b, hg_f, bars + 0, 0);
    // xg_a = reverse(concat(x_enc, d)) @ Wi_a^T + bi_a
    {
        const int N = GG;
        const int blocks = (((TT * BB) >> 4) * (N >> 4) + 7) / 8;
        gemm_concat_kernel<<<blocks, 256, 0, stream>>>(xe_bf, XX, XX, d_bf, RR,
                                                       bWi_a, KA, bi_a, xg_bf, GG,
                                                       N, KA, 1);
    }
    // backward smoothing GRU -> a (stored re-reversed)
    gru_kernel<<<GRU_BLOCKS, 256, 0, stream>>>(xg_bf, bWh_a, bh_a, a_bf,
                                               h_f, h_b, hg_f, bars + 1, 1);
    // latent scan: prior/posterior MLPs + rsample + KL
    scan_kernel<<<SCAN_BLOCKS, 256, 0, stream>>>(
        d_bf, a_bf, eps,
        bWp1, bp1, bWp2, bp2, bWp3, bp3, bWpg, bpg,
        bWq1, bq1, bWq2, bq2, bWq3, bq3, bWqg, bqg,
        inbuf, hA, hB, obuf, zprev, z_bf, kl, bars + 2);
    // decoder + log-prob
    {
        const int blocks = (((TT * BB) >> 4) * (XX >> 4) + 7) / 8;
        decoder_logprob_kernel<<<blocks, 256, 0, stream>>>(z_bf, d_bf, bWd, bd, x, lp);
    }
    finalize_kernel<<<1, 64, 0, stream>>>(lp, kl, out);
    (void)in_sizes; (void)n_in; (void)out_size; (void)ws_size;
}
